// DTransformerLayer_27917287424228
// MI455X (gfx1250) — compile-verified, matrix operand order checked by reading, not run on hardware
//
#include <hip/hip_runtime.h>
#include <hip/hip_bf16.h>

// ---------------------------------------------------------------------------
// CDNA5 (gfx1250) DTransformer attention layer.
//  - all GEMMs: v_wmma_f32_16x16x32_bf16 (bf16 operands, fp32 accum)
//  - one-shot convert pass makes every WMMA operand K-contiguous bf16, so
//    fragment loads are 2x global_load_b128 per operand
//  - Q tile staged into LDS via TDM tensor_load_to_lds + s_wait_tensorcnt
//  - softmax / cumsum / decay / maxout / layernorm in fp32 wave32 VALU
// ---------------------------------------------------------------------------

typedef __attribute__((ext_vector_type(16))) __bf16 v16bf;
typedef __attribute__((ext_vector_type(8)))  __bf16 v8bf;
typedef __attribute__((ext_vector_type(4)))  __bf16 v4bf;
typedef __attribute__((ext_vector_type(8)))  float  v8f;
typedef __attribute__((ext_vector_type(4)))  unsigned int u32x4;
typedef __attribute__((ext_vector_type(8)))  int i32x8;
typedef __attribute__((ext_vector_type(4)))  int i32x4;

#define B_SZ   8
#define S_LEN  512
#define D_MOD  1024
#define NH     16
#define DK     64

static __device__ __forceinline__ int lane_id() { return threadIdx.x & 31; }

// ISA 7.12.2 wave32 16-bit A/B fragment: lane r=lane&15 is the row (m for A,
// n for B^T-style operands); per lane the 16 elements are two contiguous
// 8-element runs at k = kbase and k = kbase+16, kbase = 8*(lane>=16).
static __device__ __forceinline__ v16bf load_frag(const __bf16* M, int ld) {
  int lane = lane_id();
  const __bf16* p = M + (size_t)(lane & 15) * ld + (((lane >> 4) & 1) << 3);
  v8bf lo = *(const v8bf*)p;
  v8bf hi = *(const v8bf*)(p + 16);
  return __builtin_shufflevector(lo, hi, 0, 1, 2, 3, 4, 5, 6, 7, 8, 9, 10, 11,
                                 12, 13, 14, 15);
}

// Same pattern but from fp32 (LDS score tile), converting to bf16.
static __device__ __forceinline__ v16bf load_frag_cvt(const float* M, int ld) {
  int lane = lane_id();
  const float* p = M + (size_t)(lane & 15) * ld + (((lane >> 4) & 1) << 3);
  v16bf f;
#pragma unroll
  for (int j = 0; j < 8; ++j) {
    f[j] = (__bf16)p[j];
    f[j + 8] = (__bf16)p[16 + j];
  }
  return f;
}

static __device__ __forceinline__ v8f wmma_bf16(v16bf a, v16bf b, v8f c) {
  return __builtin_amdgcn_wmma_f32_16x16x32_bf16(false, a, false, b, (short)0, c,
                                                 false, false);
}

static __device__ __forceinline__ float wave_max(float x) {
#pragma unroll
  for (int d = 16; d > 0; d >>= 1) x = fmaxf(x, __shfl_xor(x, d, 32));
  return x;
}
static __device__ __forceinline__ float wave_sum(float x) {
#pragma unroll
  for (int d = 16; d > 0; d >>= 1) x += __shfl_xor(x, d, 32);
  return x;
}
static __device__ __forceinline__ float wave_incl_scan(float x) {
  int lane = lane_id();
#pragma unroll
  for (int d = 1; d < 32; d <<= 1) {
    float v = __shfl_up(x, d, 32);
    if (lane >= d) x += v;
  }
  return x;
}

// ---------------------------------------------------------------------------
// Kernel 0a: elementwise fp32 -> bf16 for query/key/values.
// grid = (BSD/2048, 3), block 256, 8 elements/thread.
// ---------------------------------------------------------------------------
__global__ __launch_bounds__(256) void cvt_kernel(
    const float* __restrict__ a, const float* __restrict__ b,
    const float* __restrict__ c, __bf16* __restrict__ oa,
    __bf16* __restrict__ ob, __bf16* __restrict__ oc) {
  int z = blockIdx.y;
  const float* src = (z == 0) ? a : (z == 1) ? b : c;
  __bf16* dst = (z == 0) ? oa : (z == 1) ? ob : oc;
  size_t i = ((size_t)blockIdx.x * 256 + threadIdx.x) * 8;
  float4 x = *(const float4*)(src + i);
  float4 y = *(const float4*)(src + i + 4);
  v8bf o = {(__bf16)x.x, (__bf16)x.y, (__bf16)x.z, (__bf16)x.w,
            (__bf16)y.x, (__bf16)y.y, (__bf16)y.z, (__bf16)y.w};
  *(v8bf*)(dst + i) = o;
}

// ---------------------------------------------------------------------------
// Kernel 0b: weight transpose+convert: W[k][n] fp32 -> WT[n][k] bf16.
// grid = (32, 32, 3), block 256 (32x8 logical), 32x32 tiles via LDS.
// ---------------------------------------------------------------------------
__global__ __launch_bounds__(256) void wtr_kernel(
    const float* __restrict__ Wq, const float* __restrict__ Wv,
    const float* __restrict__ Wo, __bf16* __restrict__ WqT,
    __bf16* __restrict__ WvT, __bf16* __restrict__ WoT) {
  __shared__ float t[32][33];
  int z = blockIdx.z;
  const float* W = (z == 0) ? Wq : (z == 1) ? Wv : Wo;
  __bf16* WT = (z == 0) ? WqT : (z == 1) ? WvT : WoT;
  int bn = blockIdx.x * 32, bk = blockIdx.y * 32;
  int tx = threadIdx.x & 31, ty = threadIdx.x >> 5;
#pragma unroll
  for (int r = 0; r < 32; r += 8)
    t[ty + r][tx] = W[(size_t)(bk + ty + r) * D_MOD + bn + tx];
  __syncthreads();
#pragma unroll
  for (int r = 0; r < 32; r += 8)
    WT[(size_t)(bn + ty + r) * D_MOD + bk + tx] = (__bf16)t[tx][ty + r];
}

// ---------------------------------------------------------------------------
// Kernel 1: Q/K/V projections.  grid = (8, 256, 3), block 256 (8 waves).
// z=0: Q = Xq@Wq+bq -> [B,h,S,dk]; z=1: K likewise; z=2: V -> [B,h,dk,S].
// ---------------------------------------------------------------------------
__global__ __launch_bounds__(256) void proj_kernel(
    const __bf16* __restrict__ Xq, const __bf16* __restrict__ Xk,
    const __bf16* __restrict__ Xv, const __bf16* __restrict__ WqT,
    const __bf16* __restrict__ WvT, const float* __restrict__ bq,
    const float* __restrict__ bv, __bf16* __restrict__ Qb,
    __bf16* __restrict__ Kb, __bf16* __restrict__ Vt) {
  int which = blockIdx.z;
  const __bf16* X = (which == 0) ? Xq : (which == 1) ? Xk : Xv;
  const __bf16* WT = (which == 2) ? WvT : WqT;
  const float* bias = (which == 2) ? bv : bq;

  int wave = threadIdx.x >> 5, lane = lane_id();
  int mt = blockIdx.y;             // 0..255
  int nt = blockIdx.x * 8 + wave;  // 0..63
  const __bf16* Arow = X + (size_t)mt * 16 * D_MOD;
  const __bf16* Brow = WT + (size_t)nt * 16 * D_MOD;

  v8f c = {};
  for (int kt = 0; kt < 32; ++kt) {
    __builtin_prefetch(Arow + (kt + 1) * 32, 0, 0);
    v16bf a = load_frag(Arow + kt * 32, D_MOD);
    v16bf b = load_frag(Brow + kt * 32, D_MOD);
    c = wmma_bf16(a, b, c);
  }
#pragma unroll
  for (int r = 0; r < 8; ++r) {
    int m = ((lane >> 4) << 3) + r, n = lane & 15;
    int row = mt * 16 + m, col = nt * 16 + n;
    float val = c[r] + bias[col];
    int b_ = row >> 9, s_ = row & 511;
    int h_ = col >> 6, dd = col & 63;
    if (which == 2)  // V stored transposed: [B,h,dk,S]
      Vt[(((size_t)b_ * NH + h_) * DK + dd) * S_LEN + s_] = (__bf16)val;
    else {
      __bf16* dst = (which == 0) ? Qb : Kb;
      dst[(((size_t)b_ * NH + h_) * S_LEN + s_) * DK + dd] = (__bf16)val;
    }
  }
}

// ---------------------------------------------------------------------------
// Kernel 2: attention core.  grid = (S/16, NH, B), block 256 (8 waves).
// ---------------------------------------------------------------------------
__global__ __launch_bounds__(256) void attn_kernel(
    const __bf16* __restrict__ Qbf, const __bf16* __restrict__ Kbf,
    const __bf16* __restrict__ Vtb, const float* __restrict__ gammas,
    float* __restrict__ scores_out, __bf16* __restrict__ cat_out) {
  __shared__ float ls[16][S_LEN];      // 32 KB score tile
  __shared__ float lo[16][DK];         // 4 KB out tile
  __shared__ __bf16 lq[16 * DK];       // 2 KB Q tile (TDM destination)

  int qt = blockIdx.x, h = blockIdx.y, b = blockIdx.z;
  int wave = threadIdx.x >> 5, lane = lane_id();
  const __bf16* Qb = Qbf + (((size_t)b * NH + h) * S_LEN + (size_t)qt * 16) * DK;
  const __bf16* Kb = Kbf + ((size_t)b * NH + h) * S_LEN * DK;
  const __bf16* Vt = Vtb + ((size_t)b * NH + h) * DK * S_LEN;

  // ---- TDM: DMA the 16x64 bf16 Q tile into LDS (wave 0 issues) -----------
  if (wave == 0) {
    unsigned int lds_addr = (unsigned int)(size_t)(void*)lq;  // LDS byte offset
    unsigned long long ga = (unsigned long long)(size_t)Qb;
    u32x4 g0 = {1u,  // count=1, gather off
                lds_addr, (unsigned int)ga,
                (unsigned int)((ga >> 32) & 0x1FFFFFFu) | (2u << 30)};  // type=2
    i32x8 g1 = {(int)(1u << 16),        // data_size=2B, wg_mask=0
                (int)(64u << 16),       // tensor_dim0 = 64 (lo16 @ [31:16])
                (int)(16u << 16),       // dim0 hi16=0, tensor_dim1 = 16
                (int)(64u << 16),       // dim1 hi16=0, tile_dim0 = 64
                (int)16,                // tile_dim1 = 16, tile_dim2 = 0
                (int)64,                // tensor_dim0_stride = 64 (lo32)
                0, 0};                  // stride hi, dim1_stride = 0
    i32x4 gz4 = {0, 0, 0, 0};
    i32x8 gz8 = {0, 0, 0, 0, 0, 0, 0, 0};
    __builtin_amdgcn_tensor_load_to_lds(g0, g1, gz4, gz4, gz8, 0);
    __builtin_amdgcn_s_wait_tensorcnt(0);
  }
  __syncthreads();

  // ---- Phase A: raw scores = (Q K^T) / sqrt(dk) into LDS -----------------
  v16bf qa0 = load_frag(lq, DK);
  v16bf qa1 = load_frag(lq + 32, DK);
  const float scale = 0.125f;  // 1/sqrt(64)
  for (int kt = wave; kt < 32; kt += 8) {
    const __bf16* Kt = Kb + (size_t)kt * 16 * DK;
    v16bf kb0 = load_frag(Kt, DK);  // B^T pattern == A pattern
    v16bf kb1 = load_frag(Kt + 32, DK);
    v8f c = {};
    c = wmma_bf16(qa0, kb0, c);
    c = wmma_bf16(qa1, kb1, c);
#pragma unroll
    for (int r = 0; r < 8; ++r) {
      int m = ((lane >> 4) << 3) + r, n = lane & 15;
      ls[m][kt * 16 + n] = c[r] * scale;
    }
  }
  __syncthreads();

  // ---- Phase B: softmax -> cumsum -> distance decay -> softmax -> maxout -
  float g = -fabsf(gammas[h]);
#pragma unroll 1
  for (int rr = 0; rr < 2; ++rr) {
    int i = wave * 2 + rr;
    int qi = qt * 16 + i;  // strict causal mask: j < qi
    float sr[16], sm[16], sc[16], dcum[16], s2[16], p[16];

    float m1 = -3.4e38f;
#pragma unroll
    for (int t = 0; t < 16; ++t) {
      int j = lane + 32 * t;
      sr[t] = ls[i][j];
      sm[t] = (j < qi) ? sr[t] : -1e32f;
      m1 = fmaxf(m1, sm[t]);
    }
    m1 = wave_max(m1);
    float s1 = 0.f;
#pragma unroll
    for (int t = 0; t < 16; ++t) {
      sc[t] = __expf(sm[t] - m1);
      s1 += sc[t];
    }
    s1 = wave_sum(s1);
    float inv1 = 1.f / s1;

    float run = 0.f;
#pragma unroll 1
    for (int t = 0; t < 16; ++t) {
      float val = sc[t] * inv1;
      float pre = wave_incl_scan(val);
      dcum[t] = run + pre;
      run += __shfl(pre, 31, 32);
    }
    float total = run;  // disttotal

    float m2 = -3.4e38f;
#pragma unroll
    for (int t = 0; t < 16; ++t) {
      int j = lane + 32 * t;
      float pe = fabsf((float)(qi - j));
      float d = sqrtf(fmaxf((total - dcum[t]) * pe, 0.f));
      float eff = __expf(d * g);
      eff = fminf(fmaxf(eff, 1e-5f), 1e5f);
      s2[t] = (j < qi) ? sr[t] * eff : -1e32f;
      m2 = fmaxf(m2, s2[t]);
    }
    m2 = wave_max(m2);
    float z = 0.f;
#pragma unroll
    for (int t = 0; t < 16; ++t) {
      s2[t] = __expf(s2[t] - m2);
      z += s2[t];
    }
    z = wave_sum(z);
    float invz = 1.f / z;

    float mx = 0.f;
#pragma unroll
    for (int t = 0; t < 16; ++t) {
      int j = lane + 32 * t;
      p[t] = (j < qi) ? s2[t] * invz : 0.f;
      mx = fmaxf(mx, p[t]);
    }
    mx = wave_max(mx);
    float rs = 1.f / fmaxf(mx, 0.2f);
    float* orow = scores_out + (((size_t)b * NH + h) * S_LEN + qi) * S_LEN;
#pragma unroll
    for (int t = 0; t < 16; ++t) {
      int j = lane + 32 * t;
      p[t] *= rs;
      ls[i][j] = p[t];
      orow[j] = p[t];
    }
  }
  __syncthreads();

  // ---- Phase C: out tile = scores @ V (waves 0..3; V^T is K-contiguous) --
  if (wave < 4) {
    int nt = wave;
    v8f c = {};
    for (int kc = 0; kc < 16; ++kc) {
      v16bf a = load_frag_cvt(&ls[0][0] + kc * 32, S_LEN);
      v16bf bf = load_frag(Vt + (size_t)nt * 16 * S_LEN + kc * 32, S_LEN);
      c = wmma_bf16(a, bf, c);
    }
#pragma unroll
    for (int r = 0; r < 8; ++r) {
      int m = ((lane >> 4) << 3) + r, n = lane & 15;
      lo[m][nt * 16 + n] = c[r];
    }
  }
  __syncthreads();

  // concat [B,S,D], col = h*64+dd; vectorized 4x bf16 stores
  {
    int e = threadIdx.x;  // 256 threads, 256 chunks of 4
    int m = e >> 4, dd = (e & 15) * 4;
    v4bf o = {(__bf16)lo[m][dd], (__bf16)lo[m][dd + 1], (__bf16)lo[m][dd + 2],
              (__bf16)lo[m][dd + 3]};
    *(v4bf*)(cat_out + ((size_t)b * S_LEN + qt * 16 + m) * D_MOD + h * DK + dd) =
        o;
  }
}

// ---------------------------------------------------------------------------
// Kernel 3: x = LN(query + concat@Wo + bo).  grid = 256, block = 256.
// ---------------------------------------------------------------------------
__global__ __launch_bounds__(256) void outproj_ln_kernel(
    const __bf16* __restrict__ cat, const __bf16* __restrict__ WoT,
    const float* __restrict__ bo, const float* __restrict__ query,
    const float* __restrict__ lnw, const float* __restrict__ lnb,
    float* __restrict__ xout) {
  __shared__ float lx[16][D_MOD];  // 64 KB row tile
  int mt = blockIdx.x;
  int wave = threadIdx.x >> 5, lane = lane_id();
  const __bf16* A = cat + (size_t)mt * 16 * D_MOD;

#pragma unroll 1
  for (int nn = 0; nn < 8; ++nn) {
    int nt = wave * 8 + nn;
    const __bf16* Brow = WoT + (size_t)nt * 16 * D_MOD;
    v8f c = {};
    for (int kt = 0; kt < 32; ++kt) {
      __builtin_prefetch(Brow + (kt + 1) * 32, 0, 0);
      v16bf a = load_frag(A + kt * 32, D_MOD);
      v16bf b = load_frag(Brow + kt * 32, D_MOD);
      c = wmma_bf16(a, b, c);
    }
#pragma unroll
    for (int r = 0; r < 8; ++r) {
      int m = ((lane >> 4) << 3) + r, n = lane & 15;
      int col = nt * 16 + n;
      lx[m][col] = c[r] + bo[col] + query[((size_t)mt * 16 + m) * D_MOD + col];
    }
  }
  __syncthreads();

#pragma unroll 1
  for (int rr = 0; rr < 2; ++rr) {
    int i = wave * 2 + rr;
    float v[32], s = 0.f, ss = 0.f;
#pragma unroll
    for (int t = 0; t < 32; ++t) {
      v[t] = lx[i][lane + 32 * t];
      s += v[t];
      ss += v[t] * v[t];
    }
    s = wave_sum(s);
    ss = wave_sum(ss);
    float mu = s * (1.f / D_MOD);
    float var = ss * (1.f / D_MOD) - mu * mu;
    float rstd = rsqrtf(var + 1e-5f);
#pragma unroll
    for (int t = 0; t < 32; ++t) {
      int col = lane + 32 * t;
      xout[((size_t)mt * 16 + i) * D_MOD + col] =
          (v[t] - mu) * rstd * lnw[col] + lnb[col];
    }
  }
}

// ---------------------------------------------------------------------------
// Launch. Inputs: 0 query,1 key,2 values,3 lens(unused by reference),4 Wq,
// 5 bq,6 Wv,7 bv,8 Wo,9 bo,10 gammas,11 ln_w,12 ln_b.
// d_out = [x (8*512*1024 f32) | scores (8*16*512*512 f32)].
// Workspace (bf16): Xq|Xk|Xv (4Mi each) | WqT|WvT|WoT (1Mi each) |
//                   Qb|Kb|Vt (4Mi each) | Cat (4Mi)  = 62 MiB.
// ---------------------------------------------------------------------------
extern "C" void kernel_launch(void* const* d_in, const int* in_sizes, int n_in,
                              void* d_out, int out_size, void* d_ws,
                              size_t ws_size, hipStream_t stream) {
  const float* query = (const float*)d_in[0];
  const float* key_i = (const float*)d_in[1];
  const float* vals = (const float*)d_in[2];
  const float* Wq = (const float*)d_in[4];
  const float* bq = (const float*)d_in[5];
  const float* Wv = (const float*)d_in[6];
  const float* bv = (const float*)d_in[7];
  const float* Wo = (const float*)d_in[8];
  const float* bo = (const float*)d_in[9];
  const float* gammas = (const float*)d_in[10];
  const float* lnw = (const float*)d_in[11];
  const float* lnb = (const float*)d_in[12];

  float* xout = (float*)d_out;
  float* scores_out = xout + (size_t)B_SZ * S_LEN * D_MOD;

  const size_t BSD = (size_t)B_SZ * S_LEN * D_MOD;  // 4 Mi elements
  const size_t DD = (size_t)D_MOD * D_MOD;          // 1 Mi elements
  __bf16* p = reinterpret_cast<__bf16*>(d_ws);
  __bf16 *Xq = p, *Xk = Xq + BSD, *Xv = Xk + BSD;
  __bf16 *WqT = Xv + BSD, *WvT = WqT + DD, *WoT = WvT + DD;
  __bf16 *Qb = WoT + DD, *Kb = Qb + BSD, *Vt = Kb + BSD;
  __bf16* Cat = Vt + BSD;

  cvt_kernel<<<dim3((unsigned)(BSD / 2048), 3), 256, 0, stream>>>(
      query, key_i, vals, Xq, Xk, Xv);
  wtr_kernel<<<dim3(32, 32, 3), 256, 0, stream>>>(Wq, Wv, Wo, WqT, WvT, WoT);
  proj_kernel<<<dim3(D_MOD / 128, (B_SZ * S_LEN) / 16, 3), 256, 0, stream>>>(
      Xq, Xk, Xv, WqT, WvT, bq, bv, Qb, Kb, Vt);
  attn_kernel<<<dim3(S_LEN / 16, NH, B_SZ), 256, 0, stream>>>(
      Qb, Kb, Vt, gammas, scores_out, Cat);
  outproj_ln_kernel<<<dim3((B_SZ * S_LEN) / 16), 256, 0, stream>>>(
      Cat, WoT, bo, query, lnw, lnb, xout);
}
